// RobustNet_23845658427539
// MI455X (gfx1250) — matrix operand order repaired and checked by reading, not compiled, not run on hardware
//
#include <hip/hip_runtime.h>
#include <hip/hip_bf16.h>
#include <stdint.h>

typedef __attribute__((ext_vector_type(16))) _Float16 v16h;
typedef __attribute__((ext_vector_type(8)))  float    v8f;

#define BB 4
#define CC 32
#define TVOX 4096            // 16^3 template voxels
#define SZ   262144          // 64^3 search voxels
#define NPTS 128
#define NCHUNK 128           // S-chunks per batch
#define CHUNK (SZ / NCHUNK)  // 2048 (aligned -> fz = s>>12 constant per chunk)
#define TILE_S 128
#define LDS_PITCH 34         // 32 halves + 2 pad -> conflict-free
// logits = heat * S * log2(e); split sqrt into both f16 operands
#define ALPHA 614.9747f      // sqrt(262144 * 1.4426950408889634)

__device__ __forceinline__ float fexp2(float x) { return __builtin_amdgcn_exp2f(x); }
__device__ __forceinline__ uint32_t rol32(uint32_t v, int d) { return (v << d) | (v >> (32 - d)); }

// ---------------- threefry2x32 with key = jax.random.key(1) => (0,1) ---------
__device__ void threefry2x32_01(uint32_t x0, uint32_t x1, uint32_t* o0, uint32_t* o1) {
  const uint32_t ks0 = 0u, ks1 = 1u, ks2 = 0x1BD11BDAu ^ 0u ^ 1u;
  x0 += ks0; x1 += ks1;
  const int ra[4] = {13, 15, 26, 6};
  const int rb[4] = {17, 29, 16, 24};
#pragma unroll
  for (int i = 0; i < 4; i++) { x0 += x1; x1 = rol32(x1, ra[i]); x1 ^= x0; }
  x0 += ks1; x1 += ks2 + 1u;
#pragma unroll
  for (int i = 0; i < 4; i++) { x0 += x1; x1 = rol32(x1, rb[i]); x1 ^= x0; }
  x0 += ks2; x1 += ks0 + 2u;
#pragma unroll
  for (int i = 0; i < 4; i++) { x0 += x1; x1 = rol32(x1, ra[i]); x1 ^= x0; }
  x0 += ks0; x1 += ks1 + 3u;
#pragma unroll
  for (int i = 0; i < 4; i++) { x0 += x1; x1 = rol32(x1, rb[i]); x1 ^= x0; }
  x0 += ks1; x1 += ks2 + 4u;
#pragma unroll
  for (int i = 0; i < 4; i++) { x0 += x1; x1 = rol32(x1, ra[i]); x1 ^= x0; }
  x0 += ks2; x1 += ks0 + 5u;
  *o0 = x0; *o1 = x1;
}

// ------------- kernel 0: masked random top-128 select + gather ---------------
__global__ __launch_bounds__(256) void k_select(const float* __restrict__ tfeat,
                                                const float* __restrict__ tmask,
                                                const float* __restrict__ tlrt,
                                                _Float16* __restrict__ featsel,
                                                float* __restrict__ xyz0out) {
  __shared__ float sc[TVOX];
  __shared__ float bestv[256];
  __shared__ int   besti[256];
  __shared__ int   sel[NPTS];
  const int b = blockIdx.x, tid = threadIdx.x;

  for (int i = tid; i < TVOX; i += 256) {
    uint32_t j = (uint32_t)(b * TVOX + i);
    uint32_t c0 = (j < 8192u) ? j : (j - 8192u);
    uint32_t x0, x1;
    threefry2x32_01(c0, c0 + 8192u, &x0, &x1);
    uint32_t bits = (j < 8192u) ? x0 : x1;
    union { uint32_t u; float f; } cv;
    cv.u = (bits >> 9) | 0x3F800000u;
    float u = cv.f - 1.0f;
    sc[i] = (tmask[b * TVOX + i] > 0.0f) ? u : -1.0f;
  }
  __syncthreads();

  for (int p = 0; p < NPTS; p++) {
    float bv = -1e30f; int bi = TVOX;
    for (int i = tid; i < TVOX; i += 256) {
      float v = sc[i];
      if (v > bv || (v == bv && i < bi)) { bv = v; bi = i; }
    }
    bestv[tid] = bv; besti[tid] = bi;
    __syncthreads();
    for (int off = 128; off > 0; off >>= 1) {
      if (tid < off) {
        float v2 = bestv[tid + off]; int i2 = besti[tid + off];
        if (v2 > bestv[tid] || (v2 == bestv[tid] && i2 < besti[tid])) {
          bestv[tid] = v2; besti[tid] = i2;
        }
      }
      __syncthreads();
    }
    if (tid == 0) { sel[p] = besti[0]; sc[besti[0]] = -2.0f; }
    __syncthreads();
  }

  // gather features as f16 (pre-scaled by ALPHA), layout [b][p][c]
  for (int t = tid; t < NPTS * CC; t += 256) {
    int p = t >> 5, c = t & 31;
    int i = sel[p];
    featsel[((size_t)b * NPTS + p) * CC + c] =
        (_Float16)(tfeat[((size_t)b * CC + c) * TVOX + i] * ALPHA);
  }
  // xyz0_cam = zoom2ref(grid point, template_lrt, 16,16,16)
  if (tid < NPTS) {
    int i = sel[tid];
    float x = (float)(i & 15), y = (float)((i >> 4) & 15), z = (float)(i >> 8);
    const float* L = tlrt + b * 19;
    const float* RT = L + 3;
    float ox = (x * (1.0f / 16.0f) - 0.5f) * L[0];
    float oy = (y * (1.0f / 16.0f) - 0.5f) * L[1];
    float oz = (z * (1.0f / 16.0f) - 0.5f) * L[2];
    float* o = xyz0out + ((size_t)b * NPTS + tid) * 3;
    o[0] = RT[0] * ox + RT[1] * oy + RT[2]  * oz + RT[3];
    o[1] = RT[4] * ox + RT[5] * oy + RT[6]  * oz + RT[7];
    o[2] = RT[8] * ox + RT[9] * oy + RT[10] * oz + RT[11];
  }
}

// ------ kernel 1: WMMA f16 GEMM fused with online-softmax expectation --------
__global__ __launch_bounds__(256) void k_heat(const float* __restrict__ sfeat,
                                              const _Float16* __restrict__ featsel,
                                              float* __restrict__ partial) {
  __shared__ _Float16 tile[TILE_S * LDS_PITCH];          // [s_local][c] f16, padded
  __shared__ float red[8 * 32 * 8 * 4];                  // per-lane online states (m,s,x,y)
  const int b = blockIdx.x / NCHUNK;
  const int chunk = blockIdx.x % NCHUNK;
  const int tid = threadIdx.x;
  const int wave = tid >> 5, lane = tid & 31;
  const int g = lane >> 4, nn = lane & 15;
  const float fn = (float)nn;

  // A fragment: 16x32 f16 tile of selected features (ISA layout, 8 dwords/lane)
  union { v16h v; uint32_t u[8]; } A;
  {
    const int m = wave * 16 + nn;
    const uint32_t* fr = (const uint32_t*)(featsel + ((size_t)b * NPTS + m) * CC);
#pragma unroll
    for (int v = 0; v < 8; v++) {
      int kbase = (v & 3) * 2 + g * 8 + (v >> 2) * 16;   // lanes0-15: K{0-7,16-23}; lanes16-31: K{8-15,24-31}
      A.u[v] = fr[kbase >> 1];
    }
  }

  float mx[8], sm[8], ax[8], ay[8];
#pragma unroll
  for (int r = 0; r < 8; r++) { mx[r] = -3.0e38f; sm[r] = 0.f; ax[r] = 0.f; ay[r] = 0.f; }

  const int sBase = chunk * CHUNK;
  const float* sfb = sfeat + (size_t)b * CC * SZ;
  const int ntiles = CHUNK / TILE_S;

  for (int t = 0; t < ntiles; t++) {
    const int s0 = sBase + t * TILE_S;
    __syncthreads();
    // stage 32ch x 128s tile: batch all 16 loads, then cvt+store (pre-scale by ALPHA)
    {
      float stg[16];
#pragma unroll
      for (int q = 0; q < 16; q++) {
        int f = tid + q * 256;
        int c = f >> 7, ls = f & 127;
        stg[q] = sfb[(size_t)c * SZ + s0 + ls];
      }
#pragma unroll
      for (int q = 0; q < 16; q++) {
        int f = tid + q * 256;
        int c = f >> 7, ls = f & 127;
        tile[ls * LDS_PITCH + c] = (_Float16)(stg[q] * ALPHA);
      }
    }
    __syncthreads();

    for (int inner = 0; inner < TILE_S / 16; inner++) {
      const int sl16 = inner * 16;
      const int sl = sl16 + nn;
      union { v16h v; uint32_t u[8]; } Bf;
      const uint32_t* row = (const uint32_t*)(tile + sl * LDS_PITCH);
#pragma unroll
      for (int v = 0; v < 8; v++) Bf.u[v] = row[(v & 3) + g * 4 + (v >> 2) * 8];  // lanes0-15: K{0-7,16-23}; lanes16-31: K{8-15,24-31}

      v8f d = {};
      d = __builtin_amdgcn_wmma_f32_16x16x32_f16(false, A.v, false, Bf.v,
                                                 (short)0, d, false, false);
      const int sIter = s0 + sl16;                       // 16-aligned
      const float fx = (float)(sIter & 63) + fn;         // per-lane x coordinate
      const float fy = (float)((sIter >> 6) & 63);       // constant over the 16-tile
#pragma unroll
      for (int r = 0; r < 8; r++) {
        float h = d[r];                                  // already log2-domain logits
        float dd = h - mx[r];
        float e = fexp2(-fabsf(dd));                     // src modifiers: 1 v_exp_f32
        bool up = dd > 0.0f;
        float p = up ? 1.0f : e;
        float c = up ? e : 1.0f;
        mx[r] = up ? h : mx[r];                          // reuse vcc: 1 v_cndmask
        sm[r] = sm[r] * c + p;
        ax[r] = ax[r] * c + p * fx;
        ay[r] = ay[r] * c + p * fy;
      }
    }
  }

  // cross-lane online-softmax reduction via LDS
  __syncthreads();
#pragma unroll
  for (int r = 0; r < 8; r++) {
    int base = ((wave * 32 + lane) * 8 + r) * 4;
    red[base + 0] = mx[r]; red[base + 1] = sm[r];
    red[base + 2] = ax[r]; red[base + 3] = ay[r];
  }
  __syncthreads();
  if (tid < 128) {
    int w = tid >> 4, rl = tid & 15;
    int r = rl & 7, gg = rl >> 3;                          // D layout: M = r + 8*(lane/16)
    float M = -3.0e38f, Sm = 0.f, Sx = 0.f, Sy = 0.f;
    for (int l = 0; l < 16; l++) {
      int base = ((w * 32 + (gg * 16 + l)) * 8 + r) * 4;
      float m2 = red[base], s2 = red[base + 1];
      if (s2 == 0.0f) continue;
      if (Sm == 0.0f) { M = m2; Sm = s2; Sx = red[base + 2]; Sy = red[base + 3]; continue; }
      float nM = (M > m2) ? M : m2;
      float ca = fexp2(M - nM), cb = fexp2(m2 - nM);
      Sm = Sm * ca + s2 * cb;
      Sx = Sx * ca + red[base + 2] * cb;
      Sy = Sy * ca + red[base + 3] * cb;
      M = nM;
    }
    int row = w * 16 + rl;
    float fzc = (float)(sBase >> 12);                      // z constant per 2048-chunk
    float* o = partial + (((size_t)b * NCHUNK + chunk) * NPTS + row) * 5;
    o[0] = M; o[1] = Sm; o[2] = Sx; o[3] = Sy; o[4] = fzc * Sm;
  }
}

// --------- kernel 2: combine chunk partials -> xyz1_search (B,128,3) ---------
__global__ __launch_bounds__(256) void k_reduce(const float* __restrict__ partial,
                                                float* __restrict__ xyz1s) {
  int tid = blockIdx.x * 256 + threadIdx.x;
  if (tid >= BB * NPTS) return;
  int b = tid / NPTS, p = tid % NPTS;
  float M = -3.0e38f, Sm = 0.f, Sx = 0.f, Sy = 0.f, Sz = 0.f;
  for (int c = 0; c < NCHUNK; c++) {
    const float* q = partial + (((size_t)b * NCHUNK + c) * NPTS + p) * 5;
    float m2 = q[0], s2 = q[1];
    if (s2 == 0.0f) continue;
    if (Sm == 0.0f) { M = m2; Sm = s2; Sx = q[2]; Sy = q[3]; Sz = q[4]; continue; }
    float nM = (M > m2) ? M : m2;
    float ca = fexp2(M - nM), cb = fexp2(m2 - nM);
    Sm = Sm * ca + s2 * cb;
    Sx = Sx * ca + q[2] * cb;
    Sy = Sy * ca + q[3] * cb;
    Sz = Sz * ca + q[4] * cb;
    M = nM;
  }
  float inv = (Sm > 0.f) ? (1.0f / Sm) : 0.f;
  float* o = xyz1s + (size_t)tid * 3;
  o[0] = Sx * inv; o[1] = Sy * inv; o[2] = Sz * inv;
}

// ---------------- kernel 3: Kabsch fit + losses (tiny) -----------------------
__device__ __forceinline__ float huber(float d) {
  d = fabsf(d);
  return (d < 1.0f) ? 0.5f * d * d : d - 0.5f;
}
__device__ __forceinline__ float det3(const float R[3][3]) {
  return R[0][0] * (R[1][1] * R[2][2] - R[1][2] * R[2][1])
       - R[0][1] * (R[1][0] * R[2][2] - R[1][2] * R[2][0])
       + R[0][2] * (R[1][0] * R[2][1] - R[1][1] * R[2][0]);
}

__global__ __launch_bounds__(64) void k_final(const float* __restrict__ xyz0,
                                              const float* __restrict__ xyz1s,
                                              const float* __restrict__ slrt,
                                              const float* __restrict__ lrt01,
                                              float* __restrict__ out) {
  __shared__ float closs[BB], rloss[BB], tloss[BB];
  const int b = threadIdx.x;
  if (b < BB) {
    const float* L = slrt + b * 19;
    const float* RT = L + 3;
    float s0[3] = {0, 0, 0}, s1[3] = {0, 0, 0}, Sxy[3][3] = {{0,0,0},{0,0,0},{0,0,0}};
    for (int p = 0; p < NPTS; p++) {
      const float* a = xyz0 + ((size_t)b * NPTS + p) * 3;
      const float* w = xyz1s + ((size_t)b * NPTS + p) * 3;
      float ox = (w[0] * (1.0f / 64.0f) - 0.5f) * L[0];
      float oy = (w[1] * (1.0f / 64.0f) - 0.5f) * L[1];
      float oz = (w[2] * (1.0f / 64.0f) - 0.5f) * L[2];
      float c1v[3];
      c1v[0] = RT[0] * ox + RT[1] * oy + RT[2]  * oz + RT[3];
      c1v[1] = RT[4] * ox + RT[5] * oy + RT[6]  * oz + RT[7];
      c1v[2] = RT[8] * ox + RT[9] * oy + RT[10] * oz + RT[11];
      for (int i = 0; i < 3; i++) {
        s0[i] += a[i]; s1[i] += c1v[i];
        for (int j = 0; j < 3; j++) Sxy[i][j] += a[i] * c1v[j];
      }
    }
    const float n = (float)NPTS;
    float c0[3], c1[3];
    for (int i = 0; i < 3; i++) { c0[i] = s0[i] / n; c1[i] = s1[i] / n; }
    float H[3][3];
    for (int i = 0; i < 3; i++)
      for (int j = 0; j < 3; j++) H[i][j] = Sxy[i][j] - n * c0[i] * c1[j];

    // eigendecomposition of K = H^T H via cyclic Jacobi
    float K[3][3], V[3][3] = {{1,0,0},{0,1,0},{0,0,1}};
    for (int i = 0; i < 3; i++)
      for (int j = 0; j < 3; j++)
        K[i][j] = H[0][i] * H[0][j] + H[1][i] * H[1][j] + H[2][i] * H[2][j];
    for (int sweep = 0; sweep < 15; sweep++) {
      const int pr[3] = {0, 0, 1}, qr[3] = {1, 2, 2};
      for (int pq = 0; pq < 3; pq++) {
        int p = pr[pq], q = qr[pq];
        float apq = K[p][q];
        if (fabsf(apq) < 1e-24f) continue;
        float tau = (K[q][q] - K[p][p]) / (2.0f * apq);
        float tt = ((tau >= 0.f) ? 1.0f : -1.0f) / (fabsf(tau) + sqrtf(1.0f + tau * tau));
        float c = 1.0f / sqrtf(1.0f + tt * tt), s = tt * c;
        for (int k = 0; k < 3; k++) {
          float kp = K[k][p], kq = K[k][q];
          K[k][p] = c * kp - s * kq; K[k][q] = s * kp + c * kq;
        }
        for (int k = 0; k < 3; k++) {
          float kp = K[p][k], kq = K[q][k];
          K[p][k] = c * kp - s * kq; K[q][k] = s * kp + c * kq;
        }
        for (int k = 0; k < 3; k++) {
          float vp = V[k][p], vq = V[k][q];
          V[k][p] = c * vp - s * vq; V[k][q] = s * vp + c * vq;
        }
      }
    }
    float lam[3] = {K[0][0], K[1][1], K[2][2]};
    for (int i = 0; i < 2; i++)
      for (int j = i + 1; j < 3; j++)
        if (lam[j] > lam[i]) {
          float t2 = lam[i]; lam[i] = lam[j]; lam[j] = t2;
          for (int k = 0; k < 3; k++) { float v2 = V[k][i]; V[k][i] = V[k][j]; V[k][j] = v2; }
        }
    float U[3][3];
    for (int j = 0; j < 3; j++) {
      float sg = sqrtf(fmaxf(lam[j], 0.f));
      if (sg > 1e-20f) {
        float is = 1.0f / sg;
        for (int i = 0; i < 3; i++)
          U[i][j] = (H[i][0] * V[0][j] + H[i][1] * V[1][j] + H[i][2] * V[2][j]) * is;
      } else {
        for (int i = 0; i < 3; i++) U[i][j] = (i == j) ? 1.f : 0.f;
      }
    }
    float R[3][3];
    for (int i = 0; i < 3; i++)
      for (int k = 0; k < 3; k++)
        R[i][k] = V[i][0] * U[k][0] + V[i][1] * U[k][1] + V[i][2] * U[k][2];
    if (det3(R) < 0.f) {
      for (int i = 0; i < 3; i++) V[i][2] = -V[i][2];
      for (int i = 0; i < 3; i++)
        for (int k = 0; k < 3; k++)
          R[i][k] = V[i][0] * U[k][0] + V[i][1] * U[k][1] + V[i][2] * U[k][2];
    }
    float tv[3];
    for (int i = 0; i < 3; i++)
      tv[i] = c1[i] - (R[i][0] * c0[0] + R[i][1] * c0[1] + R[i][2] * c0[2]);
    float E[4][4] = {{R[0][0],R[0][1],R[0][2],tv[0]},
                     {R[1][0],R[1][1],R[1][2],tv[1]},
                     {R[2][0],R[2][1],R[2][2],tv[2]},
                     {0,0,0,1}};

    // g = rt1 * inv(rt0) (rt0 is rigid: affine inverse)
    const float* l0 = lrt01 + (size_t)(b * 2 + 0) * 19;
    const float* l1 = lrt01 + (size_t)(b * 2 + 1) * 19;
    float inv0[4][4] = {{0,0,0,0},{0,0,0,0},{0,0,0,0},{0,0,0,1}};
    for (int i = 0; i < 3; i++)
      for (int j = 0; j < 3; j++) inv0[i][j] = l0[3 + j * 4 + i];
    for (int i = 0; i < 3; i++) {
      float acc = 0.f;
      for (int j = 0; j < 3; j++) acc += l0[3 + j * 4 + i] * l0[3 + j * 4 + 3];
      inv0[i][3] = -acc;
    }
    float G[4][4];
    for (int i = 0; i < 4; i++)
      for (int j = 0; j < 4; j++) {
        float acc = 0.f;
        for (int k = 0; k < 4; k++) acc += l1[3 + i * 4 + k] * inv0[k][j];
        G[i][j] = acc;
      }
    // lrt_cam1_e = [lens0, (E*rt0).flat]
    out[b * 19 + 0] = l0[0]; out[b * 19 + 1] = l0[1]; out[b * 19 + 2] = l0[2];
    for (int i = 0; i < 4; i++)
      for (int j = 0; j < 4; j++) {
        float acc = 0.f;
        for (int k = 0; k < 4; k++) acc += E[i][k] * l0[3 + k * 4 + j];
        out[b * 19 + 3 + i * 4 + j] = acc;
      }
    // corner loss
    const float cxl[8] = {0.5f,0.5f,-0.5f,-0.5f,0.5f,0.5f,-0.5f,-0.5f};
    const float cyl[8] = {0.5f,0.5f,0.5f,0.5f,-0.5f,-0.5f,-0.5f,-0.5f};
    const float czl[8] = {0.5f,-0.5f,-0.5f,0.5f,0.5f,-0.5f,-0.5f,0.5f};
    float cs = 0.f;
    for (int c = 0; c < 8; c++)
      for (int i = 0; i < 3; i++) {
        float pe = E[i][0]*cxl[c] + E[i][1]*cyl[c] + E[i][2]*czl[c] + E[i][3];
        float pg = G[i][0]*cxl[c] + G[i][1]*cyl[c] + G[i][2]*czl[c] + G[i][3];
        cs += huber(pe - pg);
      }
    // rotation (euler, degrees) + translation losses
    const float r2d = 57.29577951308232f;
    float syE = sqrtf(E[0][0]*E[0][0] + E[1][0]*E[1][0]);
    float syG = sqrtf(G[0][0]*G[0][0] + G[1][0]*G[1][0]);
    float eE[3] = {atan2f(E[2][1], E[2][2]) * r2d, atan2f(-E[2][0], syE) * r2d, atan2f(E[1][0], E[0][0]) * r2d};
    float eG[3] = {atan2f(G[2][1], G[2][2]) * r2d, atan2f(-G[2][0], syG) * r2d, atan2f(G[1][0], G[0][0]) * r2d};
    float rs = huber(eE[0]-eG[0]) + huber(eE[1]-eG[1]) + huber(eE[2]-eG[2]);
    float ts = huber(E[0][3]-G[0][3]) + huber(E[1][3]-G[1][3]) + huber(E[2][3]-G[2][3]);
    closs[b] = cs; rloss[b] = rs; tloss[b] = ts;
  }
  __syncthreads();
  if (threadIdx.x == 0) {
    float cs = 0.f, rs = 0.f, ts = 0.f;
    for (int i = 0; i < BB; i++) { cs += closs[i]; rs += rloss[i]; ts += tloss[i]; }
    out[BB * 19] = cs / 96.0f + rs / 12.0f + ts / 12.0f;   // COEFF = 1
  }
}

// ------------------------------- launcher ------------------------------------
extern "C" void kernel_launch(void* const* d_in, const int* in_sizes, int n_in,
                              void* d_out, int out_size, void* d_ws, size_t ws_size,
                              hipStream_t stream) {
  (void)in_sizes; (void)n_in; (void)out_size; (void)ws_size;
  const float* tfeat = (const float*)d_in[0];   // (B,C,16,16,16)
  const float* sfeat = (const float*)d_in[1];   // (B,C,64,64,64)
  const float* tmask = (const float*)d_in[2];   // (B,16,16,16)
  const float* tlrt  = (const float*)d_in[3];   // (B,19)
  const float* slrt  = (const float*)d_in[4];   // (B,19)
  const float* lrt01 = (const float*)d_in[5];   // (B,2,19)
  float* out = (float*)d_out;

  char* ws = (char*)d_ws;
  _Float16* featsel = (_Float16*)(ws + 0);                 // B*128*32 f16     = 32768 B
  float*    xyz0    = (float*)(ws + 32768);                // B*128*3 f32      =  6144 B
  float*    partial = (float*)(ws + 38912);                // B*NCHUNK*128*5   = 1310720 B
  float*    xyz1s   = (float*)(ws + 38912 + 1310720);      // B*128*3 f32

  k_select<<<BB, 256, 0, stream>>>(tfeat, tmask, tlrt, featsel, xyz0);
  k_heat<<<BB * NCHUNK, 256, 0, stream>>>(sfeat, featsel, partial);
  k_reduce<<<(BB * NPTS + 255) / 256, 256, 0, stream>>>(partial, xyz1s);
  k_final<<<1, 64, 0, stream>>>(xyz0, xyz1s, slrt, lrt01, out);
}
